// MultiheadDilatedAttention_31507880083803
// MI455X (gfx1250) — compile-verified, matrix operand order checked by reading, not run on hardware
//
#include <hip/hip_runtime.h>

typedef __attribute__((ext_vector_type(16))) _Float16 v16h;
typedef __attribute__((ext_vector_type(2)))  _Float16 h2;
typedef __attribute__((ext_vector_type(8)))  float    v8f;

#if defined(__has_builtin)
#if __has_builtin(__builtin_amdgcn_global_load_async_to_lds_b128) && \
    __has_builtin(__builtin_amdgcn_s_wait_asynccnt)
#define USE_ASYNC_LDS 1
#endif
#endif
#ifndef USE_ASYNC_LDS
#define USE_ASYNC_LDS 0
#endif

namespace {

constexpr int kB = 2;
constexpr int kN = 8192;
constexpr int kE = 768;
constexpr int kH = 12;
constexpr int kD = 64;

#if USE_ASYNC_LDS
typedef int i4v __attribute__((vector_size(4 * sizeof(int))));
typedef __attribute__((address_space(1))) i4v gi4;  // global
typedef __attribute__((address_space(3))) i4v li4;  // LDS
#endif

// One 16-byte global -> LDS copy per lane.  Uses the CDNA5 async DMA path
// (GLOBAL_LOAD_ASYNC_TO_LDS_B128, tracked by ASYNCcnt) when the builtin is
// available, else a plain load/ds_store pair.
__device__ inline void copy16_to_lds(_Float16* lds_dst, const _Float16* gsrc) {
#if USE_ASYNC_LDS
  __builtin_amdgcn_global_load_async_to_lds_b128(
      (gi4*)(void*)gsrc, (li4*)(void*)lds_dst, 0, 0);
#else
  *reinterpret_cast<uint4*>(lds_dst) = *reinterpret_cast<const uint4*>(gsrc);
#endif
}

__device__ inline void async_fence() {
#if USE_ASYNC_LDS
  __builtin_amdgcn_s_wait_asynccnt(0);
#endif
}

// Load one 16x32 f16 WMMA fragment row-slice from a row pointer.
// A/B 16-bit layout: lanes 0-15 take half-chunks {0..7,16..23}, lanes 16-31
// take {8..15,24..31}; each 8-half chunk is one 16B (b128) load.
__device__ inline v16h ld_frag16(const _Float16* rowp, int lhalf) {
  union { v16h v; uint4 u[2]; } f;
  const uint4* p = reinterpret_cast<const uint4*>(rowp);
  f.u[0] = p[lhalf];
  f.u[1] = p[lhalf + 2];
  return f.v;
}

__global__ __launch_bounds__(256) void zero_kernel(float* p, long long n) {
  long long i = (long long)blockIdx.x * 256 + threadIdx.x;
  const long long stride = (long long)gridDim.x * 256;
  for (; i < n; i += stride) p[i] = 0.0f;
}

__global__ __launch_bounds__(256) void cvt_f32_to_f16(const float* __restrict__ src,
                                                      _Float16* __restrict__ dst, int n) {
  const int i = blockIdx.x * 256 + threadIdx.x;
  if (i < n) dst[i] = (_Float16)src[i];
}

// C = A * W^T + bias.  A is [16384 x 768] (f32 or f16), W is [768 x 768] row
// major (so B-fragment lanes read contiguous W rows).  OUT_QKV scatters f16
// into [b, h, n, d]; otherwise writes f32 row-major into d_out.
template <bool A_F16, bool OUT_QKV>
__global__ __launch_bounds__(256) void gemm_wmma_kernel(
    const float* __restrict__ Af32, const _Float16* __restrict__ Af16,
    const _Float16* __restrict__ W16, const float* __restrict__ bias,
    _Float16* __restrict__ outQKV, float* __restrict__ outF32) {
  __shared__ _Float16 As[128 * 40];
  __shared__ _Float16 Bs[128 * 40];

  const int tid = threadIdx.x;
  const int wid = tid >> 5;
  const int lane = tid & 31;
  const int lrow = lane & 15;
  const int lhalf = lane >> 4;
  const int waveM = wid & 3;   // 4 waves down  (32 rows each)
  const int waveN = wid >> 2;  // 2 waves across (64 cols each)
  const int m0 = blockIdx.x * 128;
  const int n0 = blockIdx.y * 128;

  v8f acc[2][4];
#pragma unroll
  for (int mt = 0; mt < 2; ++mt)
#pragma unroll
    for (int nt = 0; nt < 4; ++nt) acc[mt][nt] = {};

  const int ldRow = tid >> 1;
  const int ldCol = (tid & 1) * 16;

  for (int k0 = 0; k0 < kE; k0 += 32) {
    __syncthreads();
    if (A_F16) {  // pure copy -> async DMA engine
      const _Float16* src = Af16 + (size_t)(m0 + ldRow) * kE + k0 + ldCol;
      _Float16* dst = As + ldRow * 40 + ldCol;
      copy16_to_lds(dst, src);
      copy16_to_lds(dst + 8, src + 8);
    } else {      // f32 -> f16 conversion has to go through VGPRs
      const float* src = Af32 + (size_t)(m0 + ldRow) * kE + k0 + ldCol;
      union { uint4 u[2]; _Float16 h[16]; } pk;
#pragma unroll
      for (int i = 0; i < 16; ++i) pk.h[i] = (_Float16)src[i];
      uint4* dst = reinterpret_cast<uint4*>(As + ldRow * 40 + ldCol);
      dst[0] = pk.u[0]; dst[1] = pk.u[1];
    }
    {
      const _Float16* src = W16 + (size_t)(n0 + ldRow) * kE + k0 + ldCol;
      _Float16* dst = Bs + ldRow * 40 + ldCol;
      copy16_to_lds(dst, src);
      copy16_to_lds(dst + 8, src + 8);
    }
    if (k0 + 32 < kE) {  // global_prefetch_b8 for next chunk
      if (A_F16)
        __builtin_prefetch(Af16 + (size_t)(m0 + ldRow) * kE + k0 + 32 + ldCol, 0, 1);
      else
        __builtin_prefetch(Af32 + (size_t)(m0 + ldRow) * kE + k0 + 32 + ldCol, 0, 1);
      __builtin_prefetch(W16 + (size_t)(n0 + ldRow) * kE + k0 + 32 + ldCol, 0, 1);
    }
    async_fence();
    __syncthreads();

    v16h a[2], b[4];
#pragma unroll
    for (int mt = 0; mt < 2; ++mt)
      a[mt] = ld_frag16(As + (waveM * 32 + mt * 16 + lrow) * 40, lhalf);
#pragma unroll
    for (int nt = 0; nt < 4; ++nt)
      b[nt] = ld_frag16(Bs + (waveN * 64 + nt * 16 + lrow) * 40, lhalf);
#pragma unroll
    for (int mt = 0; mt < 2; ++mt)
#pragma unroll
      for (int nt = 0; nt < 4; ++nt)
        acc[mt][nt] = __builtin_amdgcn_wmma_f32_16x16x32_f16(
            false, a[mt], false, b[nt], (short)0, acc[mt][nt], false, false);
  }

#pragma unroll
  for (int nt = 0; nt < 4; ++nt) {
    const int e = n0 + waveN * 64 + nt * 16 + lrow;
    const float bvv = bias[e];
#pragma unroll
    for (int mt = 0; mt < 2; ++mt) {
#pragma unroll
      for (int j = 0; j < 8; ++j) {
        const int m = m0 + waveM * 32 + mt * 16 + j + 8 * lhalf;
        const float val = acc[mt][nt][j] + bvv;
        if (OUT_QKV) {
          const int bb = m / kN, n = m % kN;
          const int h = e / kD, dd = e % kD;
          outQKV[(((size_t)bb * kH + h) * kN + n) * kD + dd] = (_Float16)val;
        } else {
          outF32[(size_t)m * kE + e] = val;
        }
      }
    }
  }
}

// Flash-attention over one dilated instance slice.  448 blocks =
// 56 instances x 8 query-blocks of 256 rows.  8 waves/block, 32 q-rows/wave.
__global__ __launch_bounds__(256) void dilated_attn_kernel(
    const _Float16* __restrict__ Q, const _Float16* __restrict__ K,
    const _Float16* __restrict__ V, float* __restrict__ out) {
  __shared__ _Float16 Kt[32 * 72];        // [token][d + pad]
  __shared__ _Float16 Vt[64 * 48];        // transposed: [d][token + pad]
  __shared__ _Float16 Pt[8 * 32 * 48];    // per-wave P tile [qrow][ktok + pad]

  const int blk = blockIdx.x;
  const int inst = blk >> 3;
  const int qblk = blk & 7;
  int grp, rem;
  if (inst < 32)      { grp = 0; rem = inst; }
  else if (inst < 48) { grp = 1; rem = inst - 32; }
  else                { grp = 2; rem = inst - 48; }
  const int r = 1 << grp;        // dilation 1,2,4
  const int s = 2048 << grp;     // segment 2048,4096,8192
  const int off = grp;           // i % r
  const int nseg = kN / s;
  const int b = rem / (nseg * 4);
  const int rem2 = rem % (nseg * 4);
  const int seg = rem2 >> 2;
  const int h = grp * 4 + (rem2 & 3);
  const size_t headBase = (((size_t)b * kH + h) * kN) * kD;
  const int tokBase = seg * s + off;

  const int tid = threadIdx.x;
  const int wid = tid >> 5;
  const int lane = tid & 31;
  const int lrow = lane & 15;
  const int lhalf = lane >> 4;
  const int qrow0 = qblk * 256 + wid * 32;

  // Q fragments held in registers for the whole key loop (pre-scaled 1/sqrt(64))
  v16h qf[2][2];
#pragma unroll
  for (int mt = 0; mt < 2; ++mt) {
    const int j = qrow0 + mt * 16 + lrow;
    const _Float16* qp = Q + headBase + (size_t)(tokBase + j * r) * kD;
#pragma unroll
    for (int ks = 0; ks < 2; ++ks) {
      v16h f = ld_frag16(qp + ks * 32, lhalf);
#pragma unroll
      for (int i = 0; i < 16; ++i) f[i] = f[i] * (_Float16)0.125f;
      qf[mt][ks] = f;
    }
  }

  // All-ones B fragment: rs = P @ ones computes row sums of P and broadcasts
  // them to every lane of the C tile (all columns identical).
  v16h onesf;
#pragma unroll
  for (int i = 0; i < 16; ++i) onesf[i] = (_Float16)1.0f;

  v8f o_acc[2][4];
  float m_st[2][8], l_st[2][8], al_st[2][8];
#pragma unroll
  for (int mt = 0; mt < 2; ++mt) {
#pragma unroll
    for (int vt = 0; vt < 4; ++vt) o_acc[mt][vt] = {};
#pragma unroll
    for (int j = 0; j < 8; ++j) { m_st[mt][j] = -3.0e38f; l_st[mt][j] = 0.0f; }
  }

  const int ldTok = tid >> 3;
  const int ldCh = (tid & 7) * 8;
  _Float16* Pw = Pt + wid * (32 * 48);

  for (int kb = 0; kb < 64; ++kb) {
    __syncthreads();
    {  // stage 32 K rows (async DMA, row-major) and 32 V rows (transposed)
      const size_t gn = (size_t)(tokBase + (kb * 32 + ldTok) * r) * kD + ldCh;
      copy16_to_lds(Kt + ldTok * 72 + ldCh, K + headBase + gn);
      union { uint4 u; _Float16 hh[8]; } vv;
      vv.u = *reinterpret_cast<const uint4*>(V + headBase + gn);
#pragma unroll
      for (int i = 0; i < 8; ++i) Vt[(ldCh + i) * 48 + ldTok] = vv.hh[i];
    }
    async_fence();
    __syncthreads();

    // S = (Q * 1/8) . K^T   (B-fragment lanes = key tokens, contiguous rows)
    v8f sfr[2][2];
    {
      v16h bk[2][2];
#pragma unroll
      for (int nt = 0; nt < 2; ++nt)
#pragma unroll
        for (int ks = 0; ks < 2; ++ks)
          bk[nt][ks] = ld_frag16(Kt + (nt * 16 + lrow) * 72 + ks * 32, lhalf);
#pragma unroll
      for (int mt = 0; mt < 2; ++mt)
#pragma unroll
        for (int nt = 0; nt < 2; ++nt) {
          v8f c = {};
#pragma unroll
          for (int ks = 0; ks < 2; ++ks)
            c = __builtin_amdgcn_wmma_f32_16x16x32_f16(
                false, qf[mt][ks], false, bk[nt][ks], (short)0, c, false, false);
          sfr[mt][nt] = c;
        }
    }

    // Online softmax.  Row max only shifts the exponent, so a f16-accurate max
    // is plenty: pack mt=0/mt=1 partial maxima as 2xf16 in one dword and
    // butterfly both rows per shuffle (halves the ds_bpermute count).
#pragma unroll
    for (int j = 0; j < 8; ++j) {
      union { unsigned u; h2 h; } t, o;
      t.h[0] = (_Float16)fmaxf(sfr[0][0][j], sfr[0][1][j]);
      t.h[1] = (_Float16)fmaxf(sfr[1][0][j], sfr[1][1][j]);
#pragma unroll
      for (int d = 1; d < 16; d <<= 1) {
        o.u = (unsigned)__shfl_xor((int)t.u, d, 32);
        t.h[0] = t.h[0] < o.h[0] ? o.h[0] : t.h[0];
        t.h[1] = t.h[1] < o.h[1] ? o.h[1] : t.h[1];
      }
#pragma unroll
      for (int mt = 0; mt < 2; ++mt) {
        const float mo = m_st[mt][j];
        const float mn = fmaxf(mo, (float)t.h[mt]);
        const float a = __expf(mo - mn);
        m_st[mt][j] = mn;
        al_st[mt][j] = a;
        sfr[mt][0][j] = __expf(sfr[mt][0][j] - mn);
        sfr[mt][1][j] = __expf(sfr[mt][1][j] - mn);
#pragma unroll
        for (int vt = 0; vt < 4; ++vt) o_acc[mt][vt][j] *= a;
      }
    }

    // P (C-layout) -> per-wave LDS tile, re-read as A fragments for P.V
#pragma unroll
    for (int mt = 0; mt < 2; ++mt)
#pragma unroll
      for (int nt = 0; nt < 2; ++nt)
#pragma unroll
        for (int j = 0; j < 8; ++j)
          Pw[(mt * 16 + j + 8 * lhalf) * 48 + nt * 16 + lrow] =
              (_Float16)sfr[mt][nt][j];
    asm volatile("" ::: "memory");  // LDS ops are in-order per wave (DScnt)

    v16h pa[2], bvf[4];
#pragma unroll
    for (int mt = 0; mt < 2; ++mt)
      pa[mt] = ld_frag16(Pw + (mt * 16 + lrow) * 48, lhalf);
#pragma unroll
    for (int vt = 0; vt < 4; ++vt)
      bvf[vt] = ld_frag16(Vt + (vt * 16 + lrow) * 48, lhalf);

    // Row sums of P via WMMA against ones (result broadcast across lanes),
    // then the l-state update; finally O += P.V on the matrix pipe.
#pragma unroll
    for (int mt = 0; mt < 2; ++mt) {
      v8f z = {};
      const v8f rs = __builtin_amdgcn_wmma_f32_16x16x32_f16(
          false, pa[mt], false, onesf, (short)0, z, false, false);
#pragma unroll
      for (int j = 0; j < 8; ++j)
        l_st[mt][j] = l_st[mt][j] * al_st[mt][j] + rs[j];
    }
#pragma unroll
    for (int mt = 0; mt < 2; ++mt)
#pragma unroll
      for (int vt = 0; vt < 4; ++vt)
        o_acc[mt][vt] = __builtin_amdgcn_wmma_f32_16x16x32_f16(
            false, pa[mt], false, bvf[vt], (short)0, o_acc[mt][vt], false, false);
  }

  // scatter O/l into [b, n, e] f32 (instances are disjoint -> no atomics)
#pragma unroll
  for (int mt = 0; mt < 2; ++mt) {
#pragma unroll
    for (int j = 0; j < 8; ++j) {
      const int jq = qrow0 + mt * 16 + j + 8 * lhalf;
      const int n = tokBase + jq * r;
      const float inv = 1.0f / l_st[mt][j];
#pragma unroll
      for (int vt = 0; vt < 4; ++vt) {
        const int e = h * kD + vt * 16 + lrow;
        out[((size_t)b * kN + n) * kE + e] = o_acc[mt][vt][j] * inv;
      }
    }
  }
}

__global__ __launch_bounds__(256) void denom_kernel(const float* __restrict__ attn,
                                                    float* __restrict__ denom) {
  const int e = (blockIdx.x % 3) * 256 + threadIdx.x;
  const int b = (blockIdx.x / 3) & 1;
  const int nc = blockIdx.x / 6;
  float s = 0.0f;
  const int n0 = nc * 256;
  for (int n = n0; n < n0 + 256; ++n) s += attn[((size_t)b * kN + n) * kE + e];
  atomicAdd(&denom[b * kE + e], s);
}

__global__ __launch_bounds__(256) void norm_kernel(const float* __restrict__ attn,
                                                   const float* __restrict__ denom,
                                                   _Float16* __restrict__ out16) {
  const size_t i = (size_t)blockIdx.x * 256 + threadIdx.x;
  const int e = (int)(i % kE);
  const int b = (int)(i / ((size_t)kN * kE));
  out16[i] = (_Float16)(attn[i] / denom[b * kE + e]);
}

}  // namespace

extern "C" void kernel_launch(void* const* d_in, const int* in_sizes, int n_in,
                              void* d_out, int out_size, void* d_ws, size_t ws_size,
                              hipStream_t stream) {
  (void)in_sizes; (void)n_in; (void)out_size; (void)ws_size;
  const float* Xq = (const float*)d_in[0];
  const float* Xk = (const float*)d_in[1];
  const float* Xv = (const float*)d_in[2];
  const float* Wq = (const float*)d_in[3];
  const float* bq = (const float*)d_in[4];
  const float* Wk = (const float*)d_in[5];
  const float* bk = (const float*)d_in[6];
  const float* Wv = (const float*)d_in[7];
  const float* bv = (const float*)d_in[8];
  const float* Wo = (const float*)d_in[9];
  const float* bo = (const float*)d_in[10];
  float* out = (float*)d_out;

  char* ws = (char*)d_ws;
  size_t off = 0;
  auto take = [&](size_t bytes) {
    char* p = ws + off;
    off += (bytes + 255) & ~(size_t)255;
    return p;
  };
  const size_t WSZ = (size_t)kE * kE * sizeof(_Float16);
  _Float16* Wq16 = (_Float16*)take(WSZ);
  _Float16* Wk16 = (_Float16*)take(WSZ);
  _Float16* Wv16 = (_Float16*)take(WSZ);
  _Float16* Wo16 = (_Float16*)take(WSZ);
  const size_t QKV = (size_t)kB * kH * kN * kD * sizeof(_Float16);
  _Float16* Q16 = (_Float16*)take(QKV);
  _Float16* K16 = (_Float16*)take(QKV);
  _Float16* V16 = (_Float16*)take(QKV);
  float* attn = (float*)take((size_t)kB * kN * kE * sizeof(float));
  float* denom = (float*)take((size_t)kB * kE * sizeof(float));
  _Float16* attn16 = Q16;  // Q16 is dead after attention -> reuse

  zero_kernel<<<4096, 256, 0, stream>>>(attn, (long long)kB * kN * kE);
  zero_kernel<<<8, 256, 0, stream>>>(denom, (long long)kB * kE);

  const int wN = kE * kE;
  cvt_f32_to_f16<<<(wN + 255) / 256, 256, 0, stream>>>(Wq, Wq16, wN);
  cvt_f32_to_f16<<<(wN + 255) / 256, 256, 0, stream>>>(Wk, Wk16, wN);
  cvt_f32_to_f16<<<(wN + 255) / 256, 256, 0, stream>>>(Wv, Wv16, wN);
  cvt_f32_to_f16<<<(wN + 255) / 256, 256, 0, stream>>>(Wo, Wo16, wN);

  dim3 gg(128, 6, 1);  // M=16384/128, N=768/128
  gemm_wmma_kernel<false, true><<<gg, 256, 0, stream>>>(Xq, nullptr, Wq16, bq, Q16, nullptr);
  gemm_wmma_kernel<false, true><<<gg, 256, 0, stream>>>(Xk, nullptr, Wk16, bk, K16, nullptr);
  gemm_wmma_kernel<false, true><<<gg, 256, 0, stream>>>(Xv, nullptr, Wv16, bv, V16, nullptr);

  dilated_attn_kernel<<<448, 256, 0, stream>>>(Q16, K16, V16, attn);

  denom_kernel<<<192, 256, 0, stream>>>(attn, denom);
  norm_kernel<<<(kB * kN * kE) / 256, 256, 0, stream>>>(attn, denom, attn16);

  gemm_wmma_kernel<true, false><<<gg, 256, 0, stream>>>(nullptr, attn16, Wo16, bo, nullptr, out);
}
